// Decoder_30468497998160
// MI455X (gfx1250) — matrix operand order repaired and checked by reading, not compiled, not run on hardware
//
#include <hip/hip_runtime.h>
#include <hip/hip_bf16.h>
#include <math.h>

// ---------------------------------------------------------------------------
// MI455X (gfx1250) K-mode trajectory decoder.
// All dense contractions via v_wmma_f32_16x16x32_f16 (f16 A/B, f32 acc).
// f16 data path: weights pre-converted once, activations produced as f16, so
// GEMM tiles stage global->LDS as raw b128 copies (no cvt in the hot loop).
// Attention computes S^T = K_tile x Q^T so all per-tile staging is coalesced;
// P@V is split across waves with an LDS cross-wave reduction.
// Dims: H=128, N=128, T=12, K=20, KN=2560, S=4, n=32, L=640, NH=4, HD=32
// Workspace requirement: ~81 MB.
// ---------------------------------------------------------------------------

typedef __attribute__((ext_vector_type(16))) _Float16 v16h;
typedef __attribute__((ext_vector_type(8)))  float    v8f;

__device__ __forceinline__ v8f wmma16(v16h a, v16h b, v8f c) {
  return __builtin_amdgcn_wmma_f32_16x16x32_f16(false, a, false, b, (short)0, c,
                                                false, false);
}

union FragCast {
  v16h h;
  float4 f4[2];
};

// A fragment (16x32 f16) from a row-major LDS tile, lda in halfs (mult of 8).
__device__ __forceinline__ v16h load_a_frag(const _Float16* base, int lda) {
  int lane = threadIdx.x & 31;
  int r = lane & 15, g = lane >> 4;
  const _Float16* p = base + r * lda + g * 8;
  FragCast u;
  u.f4[0] = *(const float4*)p;
  u.f4[1] = *(const float4*)(p + 16);
  return u.h;
}

// B fragment (32x16 f16) from a row-major [K=32][N] LDS tile, ldb in halfs.
__device__ __forceinline__ v16h load_b_frag(const _Float16* base, int ldb) {
  int lane = threadIdx.x & 31;
  int n = lane & 15, g = lane >> 4;
  const _Float16* p = base + n;
  v16h v;
#pragma unroll
  for (int s = 0; s < 8; ++s) {
    v[s] = p[(8 * g + s) * ldb];
    v[s + 8] = p[(16 + 8 * g + s) * ldb];
  }
  return v;
}

// Stage 8 elements (16B worth of f16) into LDS; f32 source converts.
__device__ __forceinline__ void stage8(_Float16* dst, const float* src) {
  float4 a = ((const float4*)src)[0];
  float4 b = ((const float4*)src)[1];
  dst[0] = (_Float16)a.x; dst[1] = (_Float16)a.y;
  dst[2] = (_Float16)a.z; dst[3] = (_Float16)a.w;
  dst[4] = (_Float16)b.x; dst[5] = (_Float16)b.y;
  dst[6] = (_Float16)b.z; dst[7] = (_Float16)b.w;
}
__device__ __forceinline__ void stage8(_Float16* dst, const _Float16* src) {
  *(float4*)dst = *(const float4*)src;  // raw 16B copy
}

__device__ __forceinline__ float sigf(float x) {
  return 1.0f / (1.0f + __expf(-x));
}
__device__ __forceinline__ float eluf(float x) {
  return x > 0.0f ? x : (__expf(x) - 1.0f);
}

// ---------------------------------------------------------------------------
// GEMM: D[M,Nc] = A[M,K] @ B[K,Nc] (+bias) (+=C) (relu), AT in {float,f16},
// B always f16 (pre-converted weights).  C (f32) and C16 outputs nullable.
// Requires M%64==0, Nc%64==0, K%32==0.  256 threads, 64x64 tile,
// each of 8 waves computes 16x32 via two wmma accumulators.
// ---------------------------------------------------------------------------
template <typename AT>
__global__ __launch_bounds__(256) void gemm_kernel(
    const AT* __restrict__ A, const _Float16* __restrict__ B,
    const float* __restrict__ bias, float* __restrict__ C,
    _Float16* __restrict__ C16, int M, int K, int Nc, int accumulate,
    int do_relu) {
  __shared__ __align__(16) _Float16 As[64][40];
  __shared__ __align__(16) _Float16 Bs[32][72];
  int tid  = threadIdx.x;
  int wave = tid >> 5, lane = tid & 31;
  int ntiles = Nc >> 6;
  int m0 = (blockIdx.x / ntiles) << 6;
  int n0 = (blockIdx.x % ntiles) << 6;
  int wm = wave & 3, wn = wave >> 2;
  v8f acc0 = {}, acc1 = {};
  int ar = tid >> 2, ac = (tid & 3) << 3;  // 64 rows x 32 cols (A tile)
  int br = tid >> 3, bc = (tid & 7) << 3;  // 32 rows x 64 cols (B tile)
  const AT* ap = A + (size_t)(m0 + ar) * K + ac;
  const _Float16* bp = B + (size_t)br * Nc + n0 + bc;
  for (int k0 = 0; k0 < K; k0 += 32) {
    stage8(&As[ar][ac], ap + k0);
    stage8(&Bs[br][bc], bp + (size_t)k0 * Nc);
    if (k0 + 32 < K) {
      __builtin_prefetch(ap + k0 + 32, 0, 3);
      __builtin_prefetch(bp + (size_t)(k0 + 32) * Nc, 0, 3);
    }
    __syncthreads();
    v16h af  = load_a_frag(&As[wm * 16][0], 40);
    v16h bf0 = load_b_frag(&Bs[0][wn * 32], 72);
    v16h bf1 = load_b_frag(&Bs[0][wn * 32 + 16], 72);
    acc0 = wmma16(af, bf0, acc0);
    acc1 = wmma16(af, bf1, acc1);
    __syncthreads();
  }
  int g = lane >> 4, nn = lane & 15;
  int ccol = n0 + wn * 32 + nn;
  float bv0 = bias ? bias[ccol] : 0.0f;
  float bv1 = bias ? bias[ccol + 16] : 0.0f;
#pragma unroll
  for (int r = 0; r < 8; ++r) {
    int row = m0 + wm * 16 + r + 8 * g;
    float v0 = acc0[r] + bv0;
    float v1 = acc1[r] + bv1;
    if (C) {
      float* cp = C + (size_t)row * Nc + ccol;
      if (accumulate) { v0 += cp[0]; v1 += cp[16]; }
      if (do_relu)    { v0 = fmaxf(v0, 0.0f); v1 = fmaxf(v1, 0.0f); }
      cp[0]  = v0;
      cp[16] = v1;
    } else if (do_relu) {
      v0 = fmaxf(v0, 0.0f);
      v1 = fmaxf(v1, 0.0f);
    }
    if (C16) {
      _Float16* hp = C16 + (size_t)row * Nc + ccol;
      hp[0]  = (_Float16)v0;
      hp[16] = (_Float16)v1;
    }
  }
}

// Weight pre-conversion f32 -> f16.
__global__ __launch_bounds__(256) void cvt_f16_kernel(
    const float* __restrict__ src, _Float16* __restrict__ dst, int n) {
  int i = blockIdx.x * 256 + threadIdx.x;
  if (i < n) dst[i] = (_Float16)src[i];
}

// ---------------------------------------------------------------------------
// Stage A epilogue: LN over 2560 + ReLU + scatter to mode-major f16 layout
// ge16[t, k*128 + agent, h] = act[t, agent, k*128 + h].  1 block per row.
// ---------------------------------------------------------------------------
__global__ __launch_bounds__(256) void mhp_ln_kernel(
    const float* __restrict__ G0, const float* __restrict__ g,
    const float* __restrict__ beta, _Float16* __restrict__ ge16) {
  __shared__ float red[256];
  int row = blockIdx.x;  // t*128 + agent
  int t = row >> 7, nA = row & 127;
  int j = threadIdx.x;
  const float* x = G0 + (size_t)row * 2560;
  float s = 0.0f, s2 = 0.0f;
  for (int c = j; c < 2560; c += 256) {
    float v = x[c];
    s += v; s2 += v * v;
  }
  red[j] = s;
  __syncthreads();
  for (int off = 128; off > 0; off >>= 1) {
    if (j < off) red[j] += red[j + off];
    __syncthreads();
  }
  float mean = red[0] * (1.0f / 2560.0f);
  __syncthreads();
  red[j] = s2;
  __syncthreads();
  for (int off = 128; off > 0; off >>= 1) {
    if (j < off) red[j] += red[j + off];
    __syncthreads();
  }
  float var  = red[0] * (1.0f / 2560.0f) - mean * mean;
  float rstd = rsqrtf(var + 1e-5f);
  for (int c = j; c < 2560; c += 256) {
    float v = (x[c] - mean) * rstd * g[c] + beta[c];
    v = fmaxf(v, 0.0f);
    int k = c >> 7, hh = c & 127;
    ge16[((size_t)t * 2560 + k * 128 + nA) * 128 + hh] = (_Float16)v;
  }
}

// h = tile(hidden_state,(K,1)) (+f16 copy); c = tile(cn,(K,1)).
__global__ __launch_bounds__(256) void tile_state_kernel(
    const float* __restrict__ hs, const float* __restrict__ cs,
    float* __restrict__ h, _Float16* __restrict__ h16, float* __restrict__ c) {
  int idx = blockIdx.x * 256 + threadIdx.x;
  float hv = hs[idx & 16383];
  h[idx] = hv;
  h16[idx] = (_Float16)hv;
  c[idx] = cs[idx & 16383];
}

// LSTM gate elementwise.  G = [2560,512] pre-activations.  Grid 1280x256.
__global__ __launch_bounds__(256) void lstm_gates_kernel(
    const float* __restrict__ G, float* __restrict__ h,
    _Float16* __restrict__ h16, float* __restrict__ c,
    _Float16* __restrict__ out16) {
  int idx = blockIdx.x * 256 + threadIdx.x;
  int row = idx >> 7, j = idx & 127;
  const float* gp = G + ((size_t)row << 9);
  float gi = gp[j], gf = gp[j + 128], gg = gp[j + 256], go = gp[j + 384];
  float cc = c[idx];
  float c2 = sigf(gf) * cc + sigf(gi) * tanhf(gg);
  float h2 = sigf(go) * tanhf(c2);
  c[idx] = c2;
  h[idx] = h2;
  h16[idx] = (_Float16)h2;
  out16[idx] = (_Float16)h2;
}

// LSTM2 pre-step: c += social; h += tanh(c).  Grid 1280x256.
__global__ __launch_bounds__(256) void addsoc_kernel(
    float* __restrict__ h, _Float16* __restrict__ h16, float* __restrict__ c,
    const float* __restrict__ soc) {
  int idx = blockIdx.x * 256 + threadIdx.x;
  float cc = c[idx] + soc[idx];
  c[idx] = cc;
  float hn = h[idx] + tanhf(cc);
  h[idx] = hn;
  h16[idx] = (_Float16)hn;
}

// ---------------------------------------------------------------------------
// Head epilogue: y = relu(LN(H1_row)); out = y@W2 + b2 (W2:[128,2]),
// optional elu(x)+1.001.  Output scattered to [K,N,T,2].  1 block per row.
// ---------------------------------------------------------------------------
__global__ __launch_bounds__(128) void head_ep_kernel(
    const float* __restrict__ H1, const float* __restrict__ g,
    const float* __restrict__ beta, const float* __restrict__ W2,
    const float* __restrict__ b2, float* __restrict__ outp, int is_scale) {
  __shared__ float red[128];
  int row = blockIdx.x, j = threadIdx.x;
  float x = H1[(size_t)row * 128 + j];
  red[j] = x;
  __syncthreads();
  for (int off = 64; off > 0; off >>= 1) {
    if (j < off) red[j] += red[j + off];
    __syncthreads();
  }
  float mean = red[0] * (1.0f / 128.0f);
  __syncthreads();
  float d = x - mean;
  red[j] = d * d;
  __syncthreads();
  for (int off = 64; off > 0; off >>= 1) {
    if (j < off) red[j] += red[j + off];
    __syncthreads();
  }
  float rstd = rsqrtf(red[0] * (1.0f / 128.0f) + 1e-5f);
  __syncthreads();
  float y = fmaxf(d * rstd * g[j] + beta[j], 0.0f);
  red[j] = y * W2[j * 2];
  __syncthreads();
  for (int off = 64; off > 0; off >>= 1) {
    if (j < off) red[j] += red[j + off];
    __syncthreads();
  }
  float o0 = red[0] + b2[0];
  __syncthreads();
  red[j] = y * W2[j * 2 + 1];
  __syncthreads();
  for (int off = 64; off > 0; off >>= 1) {
    if (j < off) red[j] += red[j + off];
    __syncthreads();
  }
  float o1 = red[0] + b2[1];
  if (j == 0) {
    if (is_scale) { o0 = eluf(o0) + 1.001f; o1 = eluf(o1) + 1.001f; }
    int t = row / 2560, rem = row % 2560;  // rem = k*128 + agent
    size_t o = ((size_t)rem * 12 + t) * 2;
    outp[o]     = o0;
    outp[o + 1] = o1;
  }
}

// ---------------------------------------------------------------------------
// Mask: base[s][t][a][b] = any_k(|dx|<10 && |dy|<10) on scaled loc1.
// Reads loc1 from d_out section 0 ([K,N,T,2]).  Grid 48 blocks.
// ---------------------------------------------------------------------------
__global__ __launch_bounds__(256) void mask_kernel(
    const float* __restrict__ loc1, const float* __restrict__ maxv,
    const float* __restrict__ shiftv, unsigned char* __restrict__ maskb) {
  int b = blockIdx.x;  // s*12 + t
  int s = b / 12, t = b % 12;
  for (int p = threadIdx.x; p < 1024; p += 256) {
    int a = p >> 5, bb = p & 31;
    int gA = s * 32 + a, gB = s * 32 + bb;
    float mxa0 = maxv[gA * 2], mxa1 = maxv[gA * 2 + 1];
    float mxb0 = maxv[gB * 2], mxb1 = maxv[gB * 2 + 1];
    float sva0 = shiftv[gA * 2], sva1 = shiftv[gA * 2 + 1];
    float svb0 = shiftv[gB * 2], svb1 = shiftv[gB * 2 + 1];
    unsigned char m = 0;
    for (int k = 0; k < 20 && !m; ++k) {
      size_t ia = ((size_t)(k * 128 + gA) * 12 + t) * 2;
      size_t ib = ((size_t)(k * 128 + gB) * 12 + t) * 2;
      float xa = loc1[ia] * mxa0 + sva0;
      float ya = loc1[ia + 1] * mxa1 + sva1;
      float xb = loc1[ib] * mxb0 + svb0;
      float yb = loc1[ib + 1] * mxb1 + svb1;
      if (fabsf(xa - xb) < 10.0f && fabsf(ya - yb) < 10.0f) m = 1;
    }
    maskb[b * 1024 + p] = m;
  }
}

// ---------------------------------------------------------------------------
// Attention: one block per (s, t, head, 16-query tile).  L=640, HD=32.
// f16 Q/K/V.  S^T = K_tile x Q^T via WMMA (Q^T built once per block; K tiles
// staged with coalesced b128 copies).  Masked softmax in LDS.  P@V split
// across waves (wave-private V tiles) + LDS cross-wave reduction.
// Grid = 4*12*4*40 = 7680 blocks.
// ---------------------------------------------------------------------------
__global__ __launch_bounds__(256) void attn_kernel(
    const _Float16* __restrict__ Q, const _Float16* __restrict__ Kmat,
    const _Float16* __restrict__ V, const unsigned char* __restrict__ maskb,
    _Float16* __restrict__ CTX16) {
  __shared__ __align__(16) _Float16 Qt[32][24];     // Q^T (d x q), B operand
  __shared__ __align__(16) _Float16 Ka[8][16][40];  // per-wave K tile (A op)
  __shared__ float Sc[16][648];                     // scores / exp / partials
  __shared__ __align__(16) _Float16 P[16][648];     // probabilities (f16)
  __shared__ __align__(16) _Float16 Vs[8][32][40];  // per-wave V tile (B op)
  __shared__ float redA[16][16];
  __shared__ float rowM[16], rowS[16];

  int blk = blockIdx.x;
  int qt  = blk % 40;
  int sth = blk / 40;
  int hd  = sth & 3;
  int st  = sth >> 2;
  int t   = st % 12;
  int s   = st / 12;
  int tid = threadIdx.x, lane = tid & 31, wave = tid >> 5;
  const float rscale = 0.1767766952966369f;  // 1/sqrt(32)

  // Build Q^T tile [32 d][16 q], pre-scaled by 1/sqrt(HD).  Once per block.
  for (int idx = tid; idx < 512; idx += 256) {
    int q = idx >> 5, d = idx & 31;
    int i = qt * 16 + q;
    size_t row = (size_t)t * 2560 + (i >> 5) * 128 + s * 32 + (i & 31);
    Qt[d][q] = (_Float16)((float)Q[row * 128 + hd * 32 + d] * rscale);
  }
  __syncthreads();
  v16h bq = load_b_frag(&Qt[0][0], 24);  // constant across key tiles
  const unsigned char* mrow = maskb + (size_t)(s * 12 + t) * 1024;

  // scores (transposed): each wave handles key tiles kt2 = wave, wave+8, ...
  for (int kt2 = wave; kt2 < 40; kt2 += 8) {
    int krow = lane >> 1, part = lane & 1;  // 2 lanes per key row, 32B each
    int i = kt2 * 16 + krow;
    size_t row = (size_t)t * 2560 + (i >> 5) * 128 + s * 32 + (i & 31);
    const float4* src =
        (const float4*)(Kmat + row * 128 + hd * 32 + part * 16);
    float4* dst = (float4*)&Ka[wave][krow][part * 16];
    dst[0] = src[0];
    dst[1] = src[1];
    // per-wave LDS ops are in-order; wmma fragment reads this wave's tile
    v16h ak = load_a_frag(&Ka[wave][0][0], 40);
    v8f sc = {};
    sc = wmma16(ak, bq, sc);
    // C[m = key-local, n = query-local]; mask is symmetric in (i,j)
    int g = lane >> 4, qn = lane & 15;
    int iq = qt * 16 + qn;
    int jb = kt2 * 16 + 8 * g;
#pragma unroll
    for (int r = 0; r < 8; ++r) {
      unsigned char mk = mrow[(iq & 31) * 32 + ((jb + r) & 31)];
      Sc[qn][jb + r] = mk ? sc[r] : -1e8f;  // 8 consecutive floats per lane
    }
  }
  __syncthreads();

  // masked softmax over 640 keys, 16 threads per row
  int rrow = tid >> 4, seg = tid & 15;
  float lmax = -3.4e38f;
  for (int c = seg; c < 640; c += 16) lmax = fmaxf(lmax, Sc[rrow][c]);
  redA[rrow][seg] = lmax;
  __syncthreads();
  if (seg == 0) {
    float m = -3.4e38f;
    for (int i2 = 0; i2 < 16; ++i2) m = fmaxf(m, redA[rrow][i2]);
    rowM[rrow] = m;
  }
  __syncthreads();
  float mx = rowM[rrow];
  float ls = 0.0f;
  for (int c = seg; c < 640; c += 16) {
    float e = __expf(Sc[rrow][c] - mx);
    Sc[rrow][c] = e;
    ls += e;
  }
  redA[rrow][seg] = ls;
  __syncthreads();
  if (seg == 0) {
    float sm = 0.0f;
    for (int i2 = 0; i2 < 16; ++i2) sm += redA[rrow][i2];
    rowS[rrow] = sm;
  }
  __syncthreads();
  float inv = 1.0f / rowS[rrow];
  for (int c = seg; c < 640; c += 16)
    P[rrow][c] = (_Float16)(Sc[rrow][c] * inv);
  __syncthreads();  // P complete; Sc (exp values) dead after this point

  // ctx partials: each wave handles K-chunks kt2 = wave, wave+8, wave+16
  v8f ctx0 = {}, ctx1 = {};
  for (int kt2 = wave; kt2 < 20; kt2 += 8) {
    int kc = kt2 * 32;
    int i = kc + lane;  // one key row per lane: 16 halfs = 2x16B
    size_t row = (size_t)t * 2560 + (i >> 5) * 128 + s * 32 + (i & 31);
    const float4* src = (const float4*)(V + row * 128 + hd * 32);
    float4* dst = (float4*)&Vs[wave][lane][0];
    dst[0] = src[0];
    dst[1] = src[1];
    v16h ap = load_a_frag(&P[0][kc], 648);
    v16h b0 = load_b_frag(&Vs[wave][0][0], 40);
    v16h b1 = load_b_frag(&Vs[wave][0][16], 40);
    ctx0 = wmma16(ap, b0, ctx0);
    ctx1 = wmma16(ap, b1, ctx1);
  }
  // cross-wave reduction of partial contexts (reuse Sc as [8][16][32] f32)
  float* parts = &Sc[0][0];
  {
    int g = lane >> 4, nn = lane & 15;
#pragma unroll
    for (int r = 0; r < 8; ++r) {
      int m = r + 8 * g;
      parts[(wave * 16 + m) * 32 + nn]      = ctx0[r];
      parts[(wave * 16 + m) * 32 + 16 + nn] = ctx1[r];
    }
  }
  __syncthreads();
  for (int idx = tid; idx < 512; idx += 256) {
    int m = idx >> 5, d = idx & 31;
    float sum = 0.0f;
#pragma unroll
    for (int w = 0; w < 8; ++w) sum += parts[(w * 16 + m) * 32 + d];
    int iq = qt * 16 + m;
    size_t row = (size_t)t * 2560 + (iq >> 5) * 128 + s * 32 + (iq & 31);
    CTX16[row * 128 + hd * 32 + d] = (_Float16)sum;
  }
}

// ---------------------------------------------------------------------------
extern "C" void kernel_launch(void* const* d_in, const int* in_sizes, int n_in,
                              void* d_out, int out_size, void* d_ws,
                              size_t ws_size, hipStream_t stream) {
  (void)in_sizes; (void)n_in; (void)out_size; (void)ws_size;
  const float* global_embed = (const float*)d_in[0];
  const float* hidden_state = (const float*)d_in[1];
  const float* cn           = (const float*)d_in[2];
  const float* shift_values = (const float*)d_in[3];
  const float* max_values   = (const float*)d_in[4];
  const float* mhp_W  = (const float*)d_in[5];
  const float* mhp_b  = (const float*)d_in[6];
  const float* mhp_g  = (const float*)d_in[7];
  const float* mhp_be = (const float*)d_in[8];
  const float* l1_Wih = (const float*)d_in[9];
  const float* l1_Whh = (const float*)d_in[10];
  const float* l1_b   = (const float*)d_in[11];
  const float* l2_Wih = (const float*)d_in[12];
  const float* l2_Whh = (const float*)d_in[13];
  const float* l2_b   = (const float*)d_in[14];
  const float* Wq = (const float*)d_in[15];
  const float* bq = (const float*)d_in[16];
  const float* Wk = (const float*)d_in[17];
  const float* bk = (const float*)d_in[18];
  const float* Wv = (const float*)d_in[19];
  const float* bv = (const float*)d_in[20];
  const float* Wo1 = (const float*)d_in[21];
  const float* bo1 = (const float*)d_in[22];
  const float* Wo2 = (const float*)d_in[23];
  const float* bo2 = (const float*)d_in[24];
  const float* loc_W1 = (const float*)d_in[25];
  const float* loc_b1 = (const float*)d_in[26];
  const float* loc_g  = (const float*)d_in[27];
  const float* loc_be = (const float*)d_in[28];
  const float* loc_W2 = (const float*)d_in[29];
  const float* loc_b2 = (const float*)d_in[30];
  const float* scl_W1 = (const float*)d_in[31];
  const float* scl_b1 = (const float*)d_in[32];
  const float* scl_g  = (const float*)d_in[33];
  const float* scl_be = (const float*)d_in[34];
  const float* scl_W2 = (const float*)d_in[35];
  const float* scl_b2 = (const float*)d_in[36];
  float* outF = (float*)d_out;

  char* base = (char*)d_ws;
  size_t off = 0;
  auto allocB = [&](size_t bytes) {
    char* p = base + off;
    off += (bytes + 255) & ~(size_t)255;
    return p;
  };
  const size_t RC = 3932160;  // 30720*128 elements
  float*    BUF1 = (float*)allocB(RC * 4);          // stage-A C / head H1
  float*    G    = (float*)allocB(1310720 * 4);     // [KN,512] gate scratch
  float*    h    = (float*)allocB(327680 * 4);
  float*    c    = (float*)allocB(327680 * 4);
  _Float16* h16  = (_Float16*)allocB(327680 * 2);
  _Float16* ge16 = (_Float16*)allocB(RC * 2);       // [T,KN,H]
  _Float16* o116 = (_Float16*)allocB(RC * 2);       // output1 (f16)
  _Float16* o216 = (_Float16*)allocB(RC * 2);       // output2 (f16)
  _Float16* Q16  = (_Float16*)allocB(RC * 2);
  _Float16* K16  = (_Float16*)allocB(RC * 2);
  _Float16* V16  = (_Float16*)allocB(RC * 2);
  _Float16* CTX16 = (_Float16*)allocB(RC * 2);
  _Float16* MID16 = Q16;         // [30720,256] f16 aliases Q16+K16
  float*    SOC   = (float*)V16; // [T,KN,H] f32 aliases V16+CTX16
  // f16 weight arena
  _Float16* mhp_W16 = (_Float16*)allocB(327680 * 2);
  _Float16* l1_Wih16 = (_Float16*)allocB(65536 * 2);
  _Float16* l1_Whh16 = (_Float16*)allocB(65536 * 2);
  _Float16* l2_Wih16 = (_Float16*)allocB(65536 * 2);
  _Float16* l2_Whh16 = (_Float16*)allocB(65536 * 2);
  _Float16* Wq16 = (_Float16*)allocB(16384 * 2);
  _Float16* Wk16 = (_Float16*)allocB(16384 * 2);
  _Float16* Wv16 = (_Float16*)allocB(16384 * 2);
  _Float16* Wo1_16 = (_Float16*)allocB(32768 * 2);
  _Float16* Wo2_16 = (_Float16*)allocB(32768 * 2);
  _Float16* locW1_16 = (_Float16*)allocB(16384 * 2);
  _Float16* sclW1_16 = (_Float16*)allocB(16384 * 2);
  unsigned char* maskb = (unsigned char*)allocB(49152);

  auto cvtw = [&](const float* src, _Float16* dst, int n) {
    cvt_f16_kernel<<<(n + 255) / 256, 256, 0, stream>>>(src, dst, n);
  };
  auto gemmF = [&](float* C_, _Float16* C16_, const float* A_,
                   const _Float16* B_, const float* bias_, int M, int K,
                   int Nc, int acc, int relu) {
    int grid = (M >> 6) * (Nc >> 6);
    gemm_kernel<float><<<grid, 256, 0, stream>>>(A_, B_, bias_, C_, C16_, M, K,
                                                 Nc, acc, relu);
  };
  auto gemmH = [&](float* C_, _Float16* C16_, const _Float16* A_,
                   const _Float16* B_, const float* bias_, int M, int K,
                   int Nc, int acc, int relu) {
    int grid = (M >> 6) * (Nc >> 6);
    gemm_kernel<_Float16><<<grid, 256, 0, stream>>>(A_, B_, bias_, C_, C16_, M,
                                                    K, Nc, acc, relu);
  };

  // Weight pre-conversion to f16
  cvtw(mhp_W, mhp_W16, 327680);
  cvtw(l1_Wih, l1_Wih16, 65536);
  cvtw(l1_Whh, l1_Whh16, 65536);
  cvtw(l2_Wih, l2_Wih16, 65536);
  cvtw(l2_Whh, l2_Whh16, 65536);
  cvtw(Wq, Wq16, 16384);
  cvtw(Wk, Wk16, 16384);
  cvtw(Wv, Wv16, 16384);
  cvtw(Wo1, Wo1_16, 32768);
  cvtw(Wo2, Wo2_16, 32768);
  cvtw(loc_W1, locW1_16, 16384);
  cvtw(scl_W1, sclW1_16, 16384);

  // Stage A: per-mode projection + LN + ReLU + mode-major scatter (f16)
  gemmF(BUF1, nullptr, global_embed, mhp_W16, mhp_b, 1536, 128, 2560, 0, 0);
  mhp_ln_kernel<<<1536, 256, 0, stream>>>(BUF1, mhp_g, mhp_be, ge16);
  tile_state_kernel<<<1280, 256, 0, stream>>>(hidden_state, cn, h, h16, c);

  // LSTM1 scan
  for (int t = 0; t < 12; ++t) {
    gemmH(G, nullptr, ge16 + (size_t)t * 327680, l1_Wih16, l1_b, 2560, 128,
          512, 0, 0);
    gemmH(G, nullptr, h16, l1_Whh16, nullptr, 2560, 128, 512, 1, 0);
    lstm_gates_kernel<<<1280, 256, 0, stream>>>(G, h, h16, c,
                                                o116 + (size_t)t * 327680);
  }

  // Heads on output1 -> loc1 (sec 0), scale1 (sec 1)
  gemmH(BUF1, nullptr, o116, locW1_16, loc_b1, 30720, 128, 128, 0, 0);
  head_ep_kernel<<<30720, 128, 0, stream>>>(BUF1, loc_g, loc_be, loc_W2,
                                            loc_b2, outF, 0);
  gemmH(BUF1, nullptr, o116, sclW1_16, scl_b1, 30720, 128, 128, 0, 0);
  head_ep_kernel<<<30720, 128, 0, stream>>>(BUF1, scl_g, scl_be, scl_W2,
                                            scl_b2, outF + 61440, 1);

  // Proximity masks from scaled loc1
  mask_kernel<<<48, 256, 0, stream>>>(outF, max_values, shift_values, maskb);

  // Social attention (all-f16 operands)
  gemmH(nullptr, Q16, o116, Wq16, bq, 30720, 128, 128, 0, 0);
  gemmH(nullptr, K16, o116, Wk16, bk, 30720, 128, 128, 0, 0);
  gemmH(nullptr, V16, o116, Wv16, bv, 30720, 128, 128, 0, 0);
  attn_kernel<<<7680, 256, 0, stream>>>(Q16, K16, V16, maskb, CTX16);
  gemmH(nullptr, MID16, CTX16, Wo1_16, bo1, 30720, 128, 256, 0, 1);
  gemmH(SOC, nullptr, MID16, Wo2_16, bo2, 30720, 256, 128, 0, 0);

  // LSTM2 scan (state re-tiled; c += social; h += tanh(c) before gates)
  tile_state_kernel<<<1280, 256, 0, stream>>>(hidden_state, cn, h, h16, c);
  for (int t = 0; t < 12; ++t) {
    addsoc_kernel<<<1280, 256, 0, stream>>>(h, h16, c,
                                            SOC + (size_t)t * 327680);
    gemmH(G, nullptr, ge16 + (size_t)t * 327680, l2_Wih16, l2_b, 2560, 128,
          512, 0, 0);
    gemmH(G, nullptr, h16, l2_Whh16, nullptr, 2560, 128, 512, 1, 0);
    lstm_gates_kernel<<<1280, 256, 0, stream>>>(G, h, h16, c,
                                                o216 + (size_t)t * 327680);
  }

  // Heads on output2 -> loc2 (sec 2), scale2 (sec 3)
  gemmH(BUF1, nullptr, o216, locW1_16, loc_b1, 30720, 128, 128, 0, 0);
  head_ep_kernel<<<30720, 128, 0, stream>>>(BUF1, loc_g, loc_be, loc_W2,
                                            loc_b2, outF + 122880, 0);
  gemmH(BUF1, nullptr, o216, sclW1_16, scl_b1, 30720, 128, 128, 0, 0);
  head_ep_kernel<<<30720, 128, 0, stream>>>(BUF1, scl_g, scl_be, scl_W2,
                                            scl_b2, outF + 184320, 1);
}
